// SimpleMHA_45329084842096
// MI455X (gfx1250) — matrix operand order repaired and checked by reading, not compiled, not run on hardware
//
#include <hip/hip_runtime.h>
#include <hip/hip_bf16.h>

// ---------------------------------------------------------------------------
// Problem sizes (fixed by the reference)
// ---------------------------------------------------------------------------
#define DMODEL 1024
#define BATCH  4
#define SEQ    2048
#define M_TOT  (BATCH * SEQ)   // 8192 rows for all projection GEMMs

typedef __bf16 v16bf __attribute__((ext_vector_type(16)));
typedef float  v8f   __attribute__((ext_vector_type(8)));

union FragAB {
    v16bf v;
    uint4 q[2];
};

__device__ __forceinline__ unsigned short f32_to_bf16(float f) {
    unsigned u = __builtin_bit_cast(unsigned, f);
    u += 0x7FFFu + ((u >> 16) & 1u);          // round-to-nearest-even
    return (unsigned short)(u >> 16);
}

// ---------------------------------------------------------------------------
// Elementwise fp32 -> bf16 conversion (for x)
// ---------------------------------------------------------------------------
__global__ void cvt_f32_bf16_kernel(const float* __restrict__ in,
                                    unsigned short* __restrict__ out, int n) {
    int i = (blockIdx.x * blockDim.x + threadIdx.x) * 4;
    if (i + 3 < n) {
        float4 f = *(const float4*)(in + i);
        ushort4 h;
        h.x = f32_to_bf16(f.x); h.y = f32_to_bf16(f.y);
        h.z = f32_to_bf16(f.z); h.w = f32_to_bf16(f.w);
        *(ushort4*)(out + i) = h;
    }
}

// ---------------------------------------------------------------------------
// W [K][N] f32 -> Wt [N][K] bf16  (so WMMA B-fragments are contiguous loads)
// ---------------------------------------------------------------------------
__global__ void transpose_cvt_kernel(const float* __restrict__ W,
                                     unsigned short* __restrict__ Wt) {
    __shared__ float tile[32][33];
    int k0 = blockIdx.x * 32, n0 = blockIdx.y * 32;
    int tx = threadIdx.x & 31, ty = threadIdx.x >> 5;   // 256 thr: ty 0..7
    for (int r = ty; r < 32; r += 8)
        tile[r][tx] = W[(k0 + r) * DMODEL + (n0 + tx)];
    __syncthreads();
    for (int r = ty; r < 32; r += 8)
        Wt[(size_t)(n0 + r) * DMODEL + (k0 + tx)] = f32_to_bf16(tile[tx][r]);
}

// ---------------------------------------------------------------------------
// GEMM:  C[M x N] = A[M x K](bf16) * Wt[N x K](bf16)^T + bias
// Wave tile 16x64 (4 WMMA column fragments), K-loop step 32.
// mode 0: bf16 row-major out   (q, k)
// mode 1: bf16 transposed out  vT[b][n][s]   (v)
// mode 2: f32 row-major out    (final projection -> d_out)
// ---------------------------------------------------------------------------
__global__ void gemm_kernel(const unsigned short* __restrict__ A,
                            const unsigned short* __restrict__ Wt,
                            const float* __restrict__ bias,
                            void* __restrict__ out, int mode) {
    const int lane = threadIdx.x & 31;
    const int wave = threadIdx.x >> 5;
    const int half = lane >> 4;       // K-half selector (A/B fragment layout)
    const int lidx = lane & 15;       // M (for A) / N (for B) index
    const int m0 = blockIdx.x * 128 + wave * 16;
    const int n0 = blockIdx.y * 64;

    const char* Arow = (const char*)(A + (size_t)(m0 + lidx) * DMODEL);
    const char* Brow[4];
    for (int f = 0; f < 4; ++f)
        Brow[f] = (const char*)(Wt + (size_t)(n0 + f * 16 + lidx) * DMODEL);

    v8f acc[4];
    const v8f vzero = {0.f, 0.f, 0.f, 0.f, 0.f, 0.f, 0.f, 0.f};
    for (int f = 0; f < 4; ++f) acc[f] = vzero;

    for (int ks = 0; ks < DMODEL / 32; ++ks) {
        FragAB a;
        const char* ap = Arow + ks * 64 + half * 32;
        a.q[0] = *(const uint4*)(ap);
        a.q[1] = *(const uint4*)(ap + 16);
        for (int f = 0; f < 4; ++f) {
            FragAB bm;
            const char* bp = Brow[f] + ks * 64 + half * 32;
            bm.q[0] = *(const uint4*)(bp);
            bm.q[1] = *(const uint4*)(bp + 16);
            acc[f] = __builtin_amdgcn_wmma_f32_16x16x32_bf16(
                false, a.v, false, bm.v, (short)0, acc[f], false, false);
        }
    }

    for (int f = 0; f < 4; ++f) {
        const int n = n0 + f * 16 + lidx;
        const float bv = bias[n];
        if (mode == 0) {
            unsigned short* o = (unsigned short*)out;
            for (int j = 0; j < 8; ++j) {
                int m = m0 + j + half * 8;
                o[(size_t)m * DMODEL + n] = f32_to_bf16(acc[f][j] + bv);
            }
        } else if (mode == 1) {
            unsigned short* o = (unsigned short*)out;
            for (int j = 0; j < 8; ++j) {
                int m = m0 + j + half * 8;
                int b = m >> 11, s = m & (SEQ - 1);
                o[((size_t)b * DMODEL + n) * SEQ + s] = f32_to_bf16(acc[f][j] + bv);
            }
        } else {
            float* o = (float*)out;
            for (int j = 0; j < 8; ++j) {
                int m = m0 + j + half * 8;
                o[(size_t)m * DMODEL + n] = acc[f][j] + bv;
            }
        }
    }
}

// ---------------------------------------------------------------------------
// Fused causal attention: one WG (8 waves) per (batch, 16 query rows).
// LDS: Qs[16][1032] bf16 (33KB) + Sc[16][2052] f32 (128.3KB) ~= 164KB.
// Q staged via GLOBAL_LOAD_ASYNC_TO_LDS_B128 (ASYNCcnt).  Scores (QK^T) via
// WMMA -> LDS f32 -> in-LDS softmax -> in-place bf16 P -> P*V via WMMA.
// Column fragments processed in pairs for 2x WMMA ILP per wave.
// ---------------------------------------------------------------------------
#define QS_STRIDE 1032                       // bf16 elems/row (16B aligned, bank-skewed)
#define SC_STRIDE 2052                       // f32 elems/row (16B aligned, bank-skewed)
#define QS_BYTES  (16 * QS_STRIDE * 2)       // 33024
#define SC_BYTES  (16 * SC_STRIDE * 4)       // 131328
#define ATTN_LDS  (QS_BYTES + SC_BYTES)      // 164352 (< 320KB WGP LDS)

__global__ void attn_kernel(const unsigned short* __restrict__ q,
                            const unsigned short* __restrict__ k,
                            const unsigned short* __restrict__ vT,
                            unsigned short* __restrict__ o) {
    extern __shared__ char smem[];
    char* ScBase = smem + QS_BYTES;

    const int r0   = blockIdx.x * 16;       // query block start
    const int b    = blockIdx.y;
    const int lane = threadIdx.x & 31;
    const int wave = threadIdx.x >> 5;
    const int half = lane >> 4;
    const int lidx = lane & 15;

    const int nkeys  = r0 + 16;                  // last valid key col + 1
    const int kround = (nkeys + 31) & ~31;       // round to K-step of 32

    // ---- stage Q block (16 x 1024 bf16) into LDS via async b128 copies ----
    {
        const char* src = (const char*)(q + (size_t)(b * SEQ + r0) * DMODEL);
        const unsigned ldsBase = (unsigned)(size_t)smem;   // LDS byte offset
        for (int c = threadIdx.x; c < 2048; c += blockDim.x) {
            int row = c >> 7, col = c & 127;     // 128 x 16B chunks per row
            unsigned ldsOff = ldsBase + (unsigned)(row * (QS_STRIDE * 2) + col * 16);
            unsigned long long ga =
                (unsigned long long)(size_t)(src + row * (DMODEL * 2) + col * 16);
            asm volatile("global_load_async_to_lds_b128 %0, %1, off"
                         :: "v"(ldsOff), "v"(ga) : "memory");
        }
        asm volatile("s_wait_asynccnt 0x0" ::: "memory");
    }
    __syncthreads();

    // ---- scores: S = Q K^T * scale, causal mask, write f32 to LDS ----
    // Fragments processed in pairs (f0 = 2fp, f1 = 2fp+1) sharing the A-frag.
    const float scale = 0.03125f;               // 1/sqrt(1024)
    const int npair = kround >> 5;              // pairs of 16-wide fragments
    for (int fp = wave; fp < npair; fp += 8) {
        const int cbase = fp * 32;
        v8f acc0 = {0.f, 0.f, 0.f, 0.f, 0.f, 0.f, 0.f, 0.f};
        v8f acc1 = {0.f, 0.f, 0.f, 0.f, 0.f, 0.f, 0.f, 0.f};
        if (cbase <= r0 + 15) {                 // pair has unmasked elements
            const char* Krow0 = (const char*)(k + (size_t)(b * SEQ + cbase + lidx) * DMODEL);
            const char* Krow1 = Krow0 + (size_t)16 * DMODEL * 2;
            for (int ks = 0; ks < DMODEL / 32; ++ks) {
                FragAB a, b0, b1;
                const char* ap = smem + lidx * (QS_STRIDE * 2) + ks * 64 + half * 32;
                a.q[0] = *(const uint4*)(ap);
                a.q[1] = *(const uint4*)(ap + 16);
                const char* bp0 = Krow0 + ks * 64 + half * 32;
                b0.q[0] = *(const uint4*)(bp0);
                b0.q[1] = *(const uint4*)(bp0 + 16);
                const char* bp1 = Krow1 + ks * 64 + half * 32;
                b1.q[0] = *(const uint4*)(bp1);
                b1.q[1] = *(const uint4*)(bp1 + 16);
                acc0 = __builtin_amdgcn_wmma_f32_16x16x32_bf16(
                    false, a.v, false, b0.v, (short)0, acc0, false, false);
                acc1 = __builtin_amdgcn_wmma_f32_16x16x32_bf16(
                    false, a.v, false, b1.v, (short)0, acc1, false, false);
            }
        }
        const int cg0 = cbase + lidx;
        const int cg1 = cg0 + 16;
        for (int j = 0; j < 8; ++j) {
            int m  = j + half * 8;
            int rg = r0 + m;
            char* rowp = ScBase + m * (SC_STRIDE * 4);
            // masked-out fragment elements become -inf regardless of acc
            *(float*)(rowp + cg0 * 4) = (cg0 > rg) ? -__builtin_inff() : acc0[j] * scale;
            *(float*)(rowp + cg1 * 4) = (cg1 > rg) ? -__builtin_inff() : acc1[j] * scale;
        }
    }
    __syncthreads();

    // ---- softmax: each wave owns 2 rows; then repack P to bf16 in place ----
    for (int rr = 0; rr < 2; ++rr) {
        const int m = wave + rr * 8;
        char* rowp = ScBase + m * (SC_STRIDE * 4);
        float mx = -__builtin_inff();
        for (int c = lane; c < kround; c += 32)
            mx = fmaxf(mx, *(const float*)(rowp + c * 4));
        for (int off = 16; off > 0; off >>= 1)
            mx = fmaxf(mx, __shfl_xor(mx, off, 32));
        float sum = 0.f;
        for (int c = lane; c < kround; c += 32) {
            float e = __expf(*(const float*)(rowp + c * 4) - mx);
            *(float*)(rowp + c * 4) = e;
            sum += e;
        }
        for (int off = 16; off > 0; off >>= 1)
            sum += __shfl_xor(sum, off, 32);
        const float inv = 1.0f / sum;
        // In-place f32 -> bf16 repack. Safe: whole wave loads elems
        // [32t,32t+32) before storing bytes covering f32 elems [16t,16t+16).
        for (int c = lane; c < kround; c += 32) {
            float p = *(const float*)(rowp + c * 4) * inv;
            *(unsigned short*)(rowp + c * 2) = f32_to_bf16(p);
        }
    }
    __syncthreads();

    // ---- O = P V : A-frags from LDS P (bf16), B-frags from vT (contiguous) ----
    // Pairs of output-column fragments share the A-fragment.
    for (int fp = wave; fp < 32; fp += 8) {
        const int dbase = fp * 32;
        const char* Vrow0 = (const char*)(vT + ((size_t)b * DMODEL + dbase + lidx) * SEQ);
        const char* Vrow1 = Vrow0 + (size_t)16 * SEQ * 2;
        v8f acc0 = {0.f, 0.f, 0.f, 0.f, 0.f, 0.f, 0.f, 0.f};
        v8f acc1 = {0.f, 0.f, 0.f, 0.f, 0.f, 0.f, 0.f, 0.f};
        for (int ks = 0; ks < (kround >> 5); ++ks) {
            FragAB a, b0, b1;
            const char* ap = ScBase + lidx * (SC_STRIDE * 4) + ks * 64 + half * 32;
            a.q[0] = *(const uint4*)(ap);
            a.q[1] = *(const uint4*)(ap + 16);
            const char* bp0 = Vrow0 + ks * 64 + half * 32;
            b0.q[0] = *(const uint4*)(bp0);
            b0.q[1] = *(const uint4*)(bp0 + 16);
            const char* bp1 = Vrow1 + ks * 64 + half * 32;
            b1.q[0] = *(const uint4*)(bp1);
            b1.q[1] = *(const uint4*)(bp1 + 16);
            acc0 = __builtin_amdgcn_wmma_f32_16x16x32_bf16(
                false, a.v, false, b0.v, (short)0, acc0, false, false);
            acc1 = __builtin_amdgcn_wmma_f32_16x16x32_bf16(
                false, a.v, false, b1.v, (short)0, acc1, false, false);
        }
        for (int j = 0; j < 8; ++j) {
            int m = j + half * 8;
            unsigned short* orow = o + (size_t)(b * SEQ + r0 + m) * DMODEL;
            orow[dbase + lidx]      = f32_to_bf16(acc0[j]);
            orow[dbase + 16 + lidx] = f32_to_bf16(acc1[j]);
        }
    }
}

// ---------------------------------------------------------------------------
// Host launcher
// ---------------------------------------------------------------------------
extern "C" void kernel_launch(void* const* d_in, const int* in_sizes, int n_in,
                              void* d_out, int out_size, void* d_ws, size_t ws_size,
                              hipStream_t stream) {
    (void)in_sizes; (void)n_in; (void)out_size; (void)ws_size;
    const float* x  = (const float*)d_in[0];
    const float* Wq = (const float*)d_in[1];
    const float* bq = (const float*)d_in[2];
    const float* Wk = (const float*)d_in[3];
    const float* bk = (const float*)d_in[4];
    const float* Wv = (const float*)d_in[5];
    const float* bv = (const float*)d_in[6];
    const float* Wo = (const float*)d_in[7];
    const float* bo = (const float*)d_in[8];

    char* ws = (char*)d_ws;
    size_t off = 0;
    auto alloc = [&](size_t bytes) -> char* {
        char* p = ws + off;
        off += (bytes + 255) & ~(size_t)255;
        return p;
    };
    const size_t actBytes = (size_t)M_TOT * DMODEL * 2;   // 16 MB each
    unsigned short* xb  = (unsigned short*)alloc(actBytes);
    unsigned short* qb  = (unsigned short*)alloc(actBytes);
    unsigned short* kb  = (unsigned short*)alloc(actBytes);
    unsigned short* vT  = (unsigned short*)alloc(actBytes);
    unsigned short* ob  = (unsigned short*)alloc(actBytes);
    const size_t wBytes = (size_t)DMODEL * DMODEL * 2;    // 2 MB each
    unsigned short* Wtq = (unsigned short*)alloc(wBytes);
    unsigned short* Wtk = (unsigned short*)alloc(wBytes);
    unsigned short* Wtv = (unsigned short*)alloc(wBytes);
    unsigned short* Wto = (unsigned short*)alloc(wBytes);

    // x -> bf16
    cvt_f32_bf16_kernel<<<(M_TOT * DMODEL) / (256 * 4), 256, 0, stream>>>(
        x, xb, M_TOT * DMODEL);

    // weights -> transposed bf16
    dim3 tgrid(DMODEL / 32, DMODEL / 32);
    transpose_cvt_kernel<<<tgrid, 256, 0, stream>>>(Wq, Wtq);
    transpose_cvt_kernel<<<tgrid, 256, 0, stream>>>(Wk, Wtk);
    transpose_cvt_kernel<<<tgrid, 256, 0, stream>>>(Wv, Wtv);
    transpose_cvt_kernel<<<tgrid, 256, 0, stream>>>(Wo, Wto);

    // projections
    dim3 ggrid(M_TOT / 128, DMODEL / 64);
    gemm_kernel<<<ggrid, 256, 0, stream>>>(xb, Wtq, bq, qb, 0);   // Q (bf16)
    gemm_kernel<<<ggrid, 256, 0, stream>>>(xb, Wtk, bk, kb, 0);   // K (bf16)
    gemm_kernel<<<ggrid, 256, 0, stream>>>(xb, Wtv, bv, vT, 1);   // V (bf16, transposed)

    // fused causal attention
    dim3 agrid(SEQ / 16, BATCH);
    attn_kernel<<<agrid, 256, ATTN_LDS, stream>>>(qb, kb, vT, ob);

    // output projection -> f32 d_out
    gemm_kernel<<<ggrid, 256, 0, stream>>>(ob, Wto, bo, (float*)d_out, 2);
}
